// GRU_with_GCC_with_shared_params_29987461660955
// MI455X (gfx1250) — matrix operand order repaired and checked by reading, not compiled, not run on hardware
//
#include <hip/hip_runtime.h>

// ---------------------------------------------------------------------------
// GRU with gated context cells (GCC), MI455X / gfx1250, wave32 + WMMA f16->f32
//   H=512, B=64, S=512, D=512, N_AUX=2
// Phase 1: LDS-staged A-tile GEMMs (X read once), f16 operands, f32 accum,
//          software-pipelined K loop (loads run one WMMA round ahead).
// Phase 2: batch-parallel sequential scan, h resident in LDS, pipelined too.
// ---------------------------------------------------------------------------

#define Hh   512
#define Bb   64
#define Ss   512
#define Dd   512
#define H3   1536
#define H2   1024
#define LDH  520   // padded f16 row length: conflict-free ds_load_b128

typedef __attribute__((ext_vector_type(16))) _Float16 v16h;
typedef __attribute__((ext_vector_type(8)))  _Float16 v8h;
typedef __attribute__((ext_vector_type(8)))  float    v8f;
typedef __attribute__((ext_vector_type(4)))  float    v4f;

// WMMA wrapper: D = A(16x32 f16) * B(32x16 f16) + C(f32)
__device__ __forceinline__ v8f wmma_f16(v16h a, v16h b, v8f c) {
  return __builtin_amdgcn_wmma_f32_16x16x32_f16(false, a, false, b, (short)0, c,
                                                false, false);
}

// A-operand from an LDS tile [16][LDH] f16.
// `base` must point at  tileBase + l16*LDH + half*16.
// half=0: K {k0..k0+15}; half=1: K {k0+16..k0+31}
__device__ __forceinline__ v16h loadA_lds(const _Float16* base, int k0) {
  v8h c0 = *(const v8h*)(base + k0);
  v8h c1 = *(const v8h*)(base + k0 + 8);
  v16h a;
#pragma unroll
  for (int i = 0; i < 8; ++i) { a[i] = c0[i]; a[8 + i] = c1[i]; }
  return a;
}

// Stage a 16x512 fp32 A tile into LDS as f16 [16][LDH].  256 threads.
__device__ __forceinline__ void stageA(const float* __restrict__ src,
                                       _Float16* dst, int tid) {
#pragma unroll
  for (int j = 0; j < 8; ++j) {
    const int idx4 = tid + j * 256;   // 0..2047 float4 slots (16 rows x 128)
    const int row  = idx4 >> 7;
    const int c4   = (idx4 & 127) * 4;
    v4f v = *(const v4f*)(src + (size_t)row * Dd + c4);
    _Float16* d = dst + row * LDH + c4;
    d[0] = (_Float16)v[0]; d[1] = (_Float16)v[1];
    d[2] = (_Float16)v[2]; d[3] = (_Float16)v[3];
  }
}

__device__ __forceinline__ float sigmoidf_(float x) {
  return 1.0f / (1.0f + __expf(-x));
}

// ---------------------------------------------------------------------------
// Kernel 0: fp32 -> f16 conversion (weights pre-pack)
// ---------------------------------------------------------------------------
__global__ void cvt_f32_to_f16(const float* __restrict__ src,
                               _Float16* __restrict__ dst, int n) {
  int i = blockIdx.x * blockDim.x + threadIdx.x;
  int stride = gridDim.x * blockDim.x;
  for (; i < n; i += stride) dst[i] = (_Float16)src[i];
}

// ---------------------------------------------------------------------------
// Pipelined 512-deep K accumulation of four 16x16 N-tiles against one LDS
// A tile.  Loads for step ks+1 are issued before the WMMAs of step ks.
// ---------------------------------------------------------------------------
__device__ __forceinline__ void kloop4(const _Float16* arow,   // LDS, lane base
                                       const _Float16* wrow,   // weights, lane base
                                       v8f acc[4]) {
  v16h a_cur = loadA_lds(arow, 0);
  v16h bc[4];
#pragma unroll
  for (int i = 0; i < 4; ++i) bc[i] = *(const v16h*)(wrow + (size_t)(i * 16) * Dd);

#pragma unroll 5
  for (int ks = 0; ks < 15; ++ks) {
    const int kn = ks * 32 + 32;
    v16h a_nxt = loadA_lds(arow, kn);
    v16h bn[4];
#pragma unroll
    for (int i = 0; i < 4; ++i)
      bn[i] = *(const v16h*)(wrow + (size_t)(i * 16) * Dd + kn);
#pragma unroll
    for (int i = 0; i < 4; ++i) acc[i] = wmma_f16(a_cur, bc[i], acc[i]);
    a_cur = a_nxt;
#pragma unroll
    for (int i = 0; i < 4; ++i) bc[i] = bn[i];
  }
#pragma unroll
  for (int i = 0; i < 4; ++i) acc[i] = wmma_f16(a_cur, bc[i], acc[i]);
}

// ---------------------------------------------------------------------------
// Kernel 1: gi = X[BS,512] * W_ih^T[512,1536] + b_ih -> f16 [BS,1536]
// One block (8 waves) = one 16-row M-tile, A staged once in LDS.
// Wave wv covers N-groups {3wv, 3wv+1, 3wv+2} of 64 cols each.  2048 blocks.
// ---------------------------------------------------------------------------
__global__ void gemm_gi(const float* __restrict__ X,
                        const _Float16* __restrict__ Wih,   // [1536,512] f16
                        const float* __restrict__ bih,      // [1536]
                        _Float16* __restrict__ gi) {        // [BS,1536] f16
  __shared__ _Float16 Ash[16 * LDH];

  const int tid  = threadIdx.x;
  const int lane = tid & 31;
  const int wv   = tid >> 5;          // 0..7
  const int l16  = lane & 15;
  const int half = lane >> 4;
  const int mrow = blockIdx.x * 16;

  stageA(X + (size_t)mrow * Dd, Ash, tid);
  __syncthreads();

  const _Float16* arow = Ash + l16 * LDH + half * 16;

  for (int gg = 0; gg < 3; ++gg) {
    const int nbase = (wv * 3 + gg) * 64;
    const _Float16* wrow = Wih + (size_t)(nbase + l16) * Dd + half * 16;
    // pull the next weight tiles toward the WGP while WMMAs drain
    __builtin_prefetch(wrow, 0, 1);
    __builtin_prefetch(wrow + 16 * Dd, 0, 1);

    v8f acc[4] = {};
    kloop4(arow, wrow, acc);

#pragma unroll
    for (int i = 0; i < 4; ++i) {
      const int col = nbase + i * 16 + l16;
      const float bi = bih[col];
#pragma unroll
      for (int r = 0; r < 8; ++r) {
        const int row = mrow + r + 8 * half;
        gi[(size_t)row * H3 + col] = (_Float16)(acc[i][r] + bi);
      }
    }
  }
}

// ---------------------------------------------------------------------------
// Kernel 2: f = 0.5*(aux0*W_fh0^T + aux1*W_fh1^T) + 0.5*(b_fh0+b_fh1)
//           -> f16 [BS,1024].  One block = one M-tile.  N-group outer
// (uniform), aux stream inner with LDS restage; accumulators static.
// ---------------------------------------------------------------------------
__global__ void gemm_f(const float* __restrict__ AUX,      // [2,BS,512] f32
                       const _Float16* __restrict__ Wfh,   // [2,1024,512] f16
                       const float* __restrict__ bfh,      // [2,1024]
                       _Float16* __restrict__ fo) {        // [BS,1024] f16
  __shared__ _Float16 Ash[16 * LDH];

  const int tid  = threadIdx.x;
  const int lane = tid & 31;
  const int wv   = tid >> 5;          // 0..7
  const int l16  = lane & 15;
  const int half = lane >> 4;
  const int mrow = blockIdx.x * 16;

  const _Float16* arow = Ash + l16 * LDH + half * 16;

  for (int gg = 0; gg < 2; ++gg) {    // uniform across block
    const int nbase = (wv * 2 + gg) * 64;
    v8f acc[4] = {};

    for (int astr = 0; astr < 2; ++astr) {
      __syncthreads();  // previous reads of Ash complete before restage
      stageA(AUX + (size_t)astr * Bb * Ss * Dd + (size_t)mrow * Dd, Ash, tid);
      __syncthreads();

      const _Float16* wrow =
          Wfh + (size_t)astr * H2 * Dd + (size_t)(nbase + l16) * Dd + half * 16;
      __builtin_prefetch(wrow, 0, 1);
      __builtin_prefetch(wrow + 16 * Dd, 0, 1);
      kloop4(arow, wrow, acc);
    }

#pragma unroll
    for (int i = 0; i < 4; ++i) {
      const int col = nbase + i * 16 + l16;
      const float bi = 0.5f * (bfh[col] + bfh[H2 + col]);
#pragma unroll
      for (int r = 0; r < 8; ++r) {
        const int row = mrow + r + 8 * half;
        fo[(size_t)row * H2 + col] = (_Float16)(0.5f * acc[i][r] + bi);
      }
    }
  }
}

// ---------------------------------------------------------------------------
// Kernel 3: sequential GRU scan. 4 blocks x 512 threads (16 waves).
// Block owns 16 batch rows; h kept in LDS (f16, ping-pong, padded rows).
// Each wave owns 2 hidden j-tiles; per step computes r/i/n gh tiles via
// software-pipelined WMMA (next K-step's A/B loads issued before WMMAs).
// ---------------------------------------------------------------------------
__global__ void gru_scan(const float* __restrict__ hx,       // [B,H]
                         const _Float16* __restrict__ gi,    // [BS,1536]
                         const _Float16* __restrict__ fo,    // [BS,1024]
                         const _Float16* __restrict__ Whh,   // [1536,512] f16
                         const float* __restrict__ bhh,      // [1536]
                         float* __restrict__ out) {          // [B*S*H + B*H]
  __shared__ _Float16 hsh[2][16 * LDH];

  const int tid  = threadIdx.x;
  const int lane = tid & 31;
  const int wv   = tid >> 5;          // 0..15
  const int l16  = lane & 15;
  const int half = lane >> 4;
  const int batchBase = blockIdx.x * 16;

  // init h0 from hx
  for (int idx = tid; idx < 16 * Hh; idx += 512) {
    const int r = idx >> 9, c = idx & (Hh - 1);
    hsh[0][r * LDH + c] = (_Float16)hx[(size_t)(batchBase + r) * Hh + c];
  }
  __syncthreads();

  // per-wave hoisted bias loads for its two j-tiles
  float brv[2], biv[2], bnv[2];
#pragma unroll
  for (int j = 0; j < 2; ++j) {
    const int jc = (wv * 2 + j) * 16 + l16;
    brv[j] = bhh[jc];
    biv[j] = bhh[Hh + jc];
    bnv[j] = bhh[2 * Hh + jc];
  }

  for (int t = 0; t < Ss; ++t) {
    const _Float16* cur = hsh[t & 1];
    _Float16* nxt = hsh[(t + 1) & 1];

#pragma unroll
    for (int j = 0; j < 2; ++j) {
      const int jt = wv * 2 + j;        // 0..31
      const _Float16* wR = Whh + (size_t)(jt * 16 + l16) * Dd + half * 16;
      const _Float16* wI = wR + (size_t)Hh * Dd;
      const _Float16* wN = wI + (size_t)Hh * Dd;
      const _Float16* hrow = cur + l16 * LDH + half * 16;

      v8f ar = {}, ai = {}, an = {};
      // software-pipelined K loop over h (LDS) x Whh (global/L2)
      v16h a_cur = loadA_lds(hrow, 0);
      v16h brc = *(const v16h*)(wR);
      v16h bic = *(const v16h*)(wI);
      v16h bnc = *(const v16h*)(wN);
#pragma unroll 5
      for (int ks = 0; ks < 15; ++ks) {
        const int kn = ks * 32 + 32;
        v16h a_nxt = loadA_lds(hrow, kn);
        v16h brn = *(const v16h*)(wR + kn);
        v16h bin = *(const v16h*)(wI + kn);
        v16h bnn = *(const v16h*)(wN + kn);
        ar = wmma_f16(a_cur, brc, ar);
        ai = wmma_f16(a_cur, bic, ai);
        an = wmma_f16(a_cur, bnc, an);
        a_cur = a_nxt; brc = brn; bic = bin; bnc = bnn;
      }
      ar = wmma_f16(a_cur, brc, ar);
      ai = wmma_f16(a_cur, bic, ai);
      an = wmma_f16(a_cur, bnc, an);

      // fused gates + state update for this 16x16 tile
      const int jc = jt * 16 + l16;     // hidden index
#pragma unroll
      for (int r = 0; r < 8; ++r) {
        const int m = r + 8 * half;                 // row in tile
        const int b = batchBase + m;                // batch index
        const size_t rowg = ((size_t)b * Ss + t);
        const float gir = (float)gi[rowg * H3 + jt * 16 + l16];
        const float gii = (float)gi[rowg * H3 + Hh + jt * 16 + l16];
        const float gin = (float)gi[rowg * H3 + 2 * Hh + jt * 16 + l16];
        const float fr  = (float)fo[rowg * H2 + jt * 16 + l16];
        const float fi  = (float)fo[rowg * H2 + Hh + jt * 16 + l16];
        const float ghr = ar[r] + brv[j];
        const float ghi = ai[r] + biv[j];
        const float ghn = an[r] + bnv[j];
        const float hprev = (float)cur[m * LDH + jc];

        const float rg = sigmoidf_(gir + ghr + fr);
        const float ig = sigmoidf_(gii + ghi + fi);
        const float ng = tanhf(gin + rg * ghn);
        const float hy = ng + ig * (hprev - ng);

        out[rowg * Hh + jc] = hy;
        nxt[m * LDH + jc] = (_Float16)hy;
      }
    }
    __syncthreads();
  }

  // final hidden state (S even -> lives in buffer 0)
  float* hfin = out + (size_t)Bb * Ss * Hh;
  for (int idx = tid; idx < 16 * Hh; idx += 512) {
    const int r = idx >> 9, c = idx & (Hh - 1);
    hfin[(size_t)(batchBase + r) * Hh + c] = (float)hsh[0][r * LDH + c];
  }
}

// ---------------------------------------------------------------------------
// launch
// ---------------------------------------------------------------------------
extern "C" void kernel_launch(void* const* d_in, const int* in_sizes, int n_in,
                              void* d_out, int out_size, void* d_ws,
                              size_t ws_size, hipStream_t stream) {
  const float* X    = (const float*)d_in[0];   // [B,S,D]
  const float* AUX  = (const float*)d_in[1];   // [2,B,S,D]
  const float* hx   = (const float*)d_in[2];   // [B,H]
  const float* Wih  = (const float*)d_in[3];   // [3H,D]
  const float* Wfh  = (const float*)d_in[4];   // [2,2H,D]
  const float* bih  = (const float*)d_in[5];   // [3H]
  const float* bfh  = (const float*)d_in[6];   // [2,2H]
  const float* Whh  = (const float*)d_in[7];   // [3H,H]
  const float* bhh  = (const float*)d_in[8];   // [3H]
  float* out = (float*)d_out;

  // workspace layout (f16 regions)
  char* ws = (char*)d_ws;
  _Float16* whh16 = (_Float16*)(ws);                        // 1536*512
  _Float16* wih16 = (_Float16*)(ws + (size_t)1572864);      // 1536*512
  _Float16* wfh16 = (_Float16*)(ws + (size_t)3145728);      // 2*1024*512
  _Float16* gi16  = (_Float16*)(ws + (size_t)5242880);      // 32768*1536
  _Float16* f16   = (_Float16*)(ws + (size_t)105906176);    // 32768*1024

  // 0) pre-pack weights to f16
  cvt_f32_to_f16<<<512, 256, 0, stream>>>(Whh, whh16, H3 * Hh);
  cvt_f32_to_f16<<<512, 256, 0, stream>>>(Wih, wih16, H3 * Dd);
  cvt_f32_to_f16<<<512, 256, 0, stream>>>(Wfh, wfh16, 2 * H2 * Dd);

  // 1) gi GEMM: one block per 16-row M-tile (A staged in LDS once)
  gemm_gi<<<2048, 256, 0, stream>>>(X, wih16, bih, gi16);

  // 2) f GEMM: one block per M-tile, both aux streams staged sequentially
  gemm_f<<<2048, 256, 0, stream>>>(AUX, wfh16, bfh, f16);

  // 3) sequential scan: 4 blocks (16 batch rows each) x 512 threads
  gru_scan<<<4, 512, 0, stream>>>(hx, gi16, f16, whh16, bhh, out);
}
